// KR_ERE_71940702208393
// MI455X (gfx1250) — matrix operand order repaired and checked by reading, not compiled, not run on hardware
//
#include <hip/hip_runtime.h>

// ---------------------------------------------------------------------------
// CDNA5 (gfx1250) implementation: all conv1d's and the attention projection
// are bf16 WMMA GEMMs (v_wmma_f32_16x16x32_bf16), f32 accumulate.
// X tiles staged into LDS via GLOBAL_LOAD_ASYNC_TO_LDS when available.
// ---------------------------------------------------------------------------

typedef __attribute__((ext_vector_type(16))) __bf16    v16bf;
typedef __attribute__((ext_vector_type(8)))  float     v8f;
typedef __attribute__((ext_vector_type(4)))  unsigned  u32x4;
typedef __attribute__((ext_vector_type(4)))  int       i32x4;

#if defined(__has_builtin)
# if __has_builtin(__builtin_amdgcn_global_load_async_to_lds_b128) && \
     __has_builtin(__builtin_amdgcn_s_wait_asynccnt)
#  define USE_ASYNC_LDS 1
# endif
#endif
#ifndef USE_ASYNC_LDS
# define USE_ASYNC_LDS 0
#endif

#if USE_ASYNC_LDS
// builtin signature (from compiler diagnostic): (v4i AS1*, v4i AS3*, Ii, Ii)
typedef __attribute__((address_space(1))) i32x4 g_i32x4;   // global
typedef __attribute__((address_space(3))) i32x4 l_i32x4;   // LDS
#endif

#define VEC   300
#define CONV  512
#define SEN   4096
#define CI_P  320            // ci padded 300 -> 320 (10 chunks of K=32)
#define XCOLS (SEN + 8)      // 4 halo columns each side
#define TILE_COLS 72         // 64 output cols + 8 halo

// ---- bf16 helpers (manual RNE, storage as unsigned short) ------------------
__device__ __forceinline__ unsigned short f2bf(float f) {
    unsigned u = __float_as_uint(f);
    u = u + 0x7FFFu + ((u >> 16) & 1u);
    return (unsigned short)(u >> 16);
}
__device__ __forceinline__ float bf2f(unsigned short h) {
    return __uint_as_float(((unsigned)h) << 16);
}

union FragU { u32x4 q[2]; v16bf v; };

// A fragment, 16x32 bf16 (M x K): lane 0-15 = row M, holds K {0..7,16..23};
// lanes 16-31 hold K {8..15,24..31}. p already includes +laneHalf*8 elements.
__device__ __forceinline__ v16bf load_frag_a(const unsigned short* p) {
    FragU u;
    u.q[0] = *(const u32x4*)(p);        // K +0..7
    u.q[1] = *(const u32x4*)(p + 16);   // K +16..23
    return u.v;
}
// B fragment, 32x16 bf16 (K x N): lanes 0-15 hold K {0..7,16..23},
// lanes 16-31 hold K {8..15,24..31}; p already includes +laneHalf*8.
__device__ __forceinline__ v16bf load_frag_b(const unsigned short* p) {
    FragU u;
    u.q[0] = *(const u32x4*)(p);
    u.q[1] = *(const u32x4*)(p + 16);
    return u.v;
}

__device__ __forceinline__ v8f wmma_bf16(v16bf a, v16bf b, v8f c) {
    return __builtin_amdgcn_wmma_f32_16x16x32_bf16(
        /*neg_a=*/false, a, /*neg_b=*/false, b,
        /*c_mod=*/(short)0, c, /*reuse_a=*/false, /*reuse_b=*/false);
}

// ---------------------------------------------------------------------------
// Prep kernels
// ---------------------------------------------------------------------------

// x [300,4096] f32 -> XB [XCOLS][CI_P] bf16 column-major (ci contiguous)
__global__ __launch_bounds__(256) void kConvertX(const float* __restrict__ x,
                                                 unsigned short* __restrict__ XB) {
    int idx = blockIdx.x * 256 + threadIdx.x;
    if (idx >= XCOLS * CI_P) return;
    int col = idx / CI_P, ci = idx % CI_P;
    int l = col - 4;
    float v = (l >= 0 && l < SEN && ci < VEC) ? x[(size_t)ci * SEN + l] : 0.f;
    XB[idx] = f2bf(v);
}

struct WPtrs { const float* p[12]; };

// W [512,300,K] f32 -> WB[conv] as [co][k][ci=CI_P] bf16
__global__ __launch_bounds__(256) void kConvertW(WPtrs wp, unsigned short* __restrict__ WB) {
    int conv = blockIdx.z;
    int b = conv % 3, g = conv / 3;
    int Kt = 3 + b;
    int elems = CONV * Kt * CI_P;
    int idx = blockIdx.x * 256 + threadIdx.x;
    if (idx >= elems) return;
    int ci = idx % CI_P;
    int t  = idx / CI_P;
    int k  = t % Kt;
    int co = t / Kt;
    const float* W = wp.p[conv];
    float v = (ci < VEC) ? W[((size_t)co * VEC + ci) * Kt + k] : 0.f;
    int sk = (b == 0) ? 0 : ((b == 1) ? 3 : 7);         // sum of prior K's
    size_t base = ((size_t)g * 12 + sk) * CONV * CI_P;
    WB[base + idx] = f2bf(v);
}

// attention_W [300,512] f32 -> AWB [304][512] bf16 (rows >=300 zero)
__global__ __launch_bounds__(256) void kConvertAttW(const float* __restrict__ aw,
                                                    unsigned short* __restrict__ AWB) {
    int idx = blockIdx.x * 256 + threadIdx.x;
    if (idx >= 304 * CONV) return;
    int m = idx / CONV, k = idx % CONV;
    AWB[idx] = f2bf(m < VEC ? aw[(size_t)m * CONV + k] : 0.f);
}

// E1 = chemW@e1 + chemB ; E2 = disW@e2 + disB   (grid = 4 blocks of 256)
__global__ __launch_bounds__(256) void kEntGemv(const float* __restrict__ chemW,
                                                const float* __restrict__ chemB,
                                                const float* __restrict__ disW,
                                                const float* __restrict__ disB,
                                                const float* __restrict__ e1,
                                                const float* __restrict__ e2,
                                                float* __restrict__ E) {
    int b = blockIdx.x;               // 0,1 -> E1 ; 2,3 -> E2
    int side = b >> 1;
    int co = (b & 1) * 256 + threadIdx.x;
    const float* W  = side ? disW : chemW;
    const float* bb = side ? disB : chemB;
    const float* ev = side ? e2 : e1;
    float acc = bb[co];
    for (int ci = 0; ci < VEC; ++ci) acc += W[(size_t)co * VEC + ci] * ev[ci];
    E[side * CONV + co] = acc;
}

// ---------------------------------------------------------------------------
// Fused conv pair + gate product:  Cout[l][co] = tanh(yT+bt)*relu(yG+bg+E)
// Block: 256 thr (8 waves). Tile M=128 (16 rows/wave), N=64.
// ---------------------------------------------------------------------------
__global__ __launch_bounds__(256)
void kConvGemm(const unsigned short* __restrict__ XB,
               const unsigned short* __restrict__ WT,
               const unsigned short* __restrict__ WG,
               const float* __restrict__ bt, const float* __restrict__ bg,
               const float* __restrict__ Evec,
               unsigned short* __restrict__ Cout, int Kt) {
    __shared__ unsigned short tile[TILE_COLS * CI_P];   // 46,080 B

    const int n0   = blockIdx.x * 64;
    const int mblk = blockIdx.y * 128;
    const int tid  = threadIdx.x;
    const int wave = tid >> 5;
    const int lane = tid & 31;
    const int l15  = lane & 15;
    const int lh   = lane >> 4;
    const int m0   = mblk + wave * 16;

    // warm L2/WGP$ for this block's weight rows (global_prefetch_b8)
    __builtin_prefetch(WT + (size_t)m0 * Kt * CI_P, 0, 1);
    __builtin_prefetch(WG + (size_t)m0 * Kt * CI_P, 0, 1);

    // stage X columns [n0, n0+72) of the padded buffer into LDS
    {
        const int QPC = CI_P / 8;                       // 40 u32x4 per column
#if USE_ASYNC_LDS
        for (int t = tid; t < TILE_COLS * QPC; t += 256) {
            int col = t / QPC, q = t % QPC;
            const u32x4* src = ((const u32x4*)(XB + (size_t)(n0 + col) * CI_P)) + q;
            u32x4*       dst = ((u32x4*)tile) + t;
            // direct global -> LDS DMA, tracked by ASYNCcnt (no VGPR staging)
            __builtin_amdgcn_global_load_async_to_lds_b128(
                (g_i32x4*)(unsigned long long)(__UINTPTR_TYPE__)src,
                (l_i32x4*)(unsigned)(__UINTPTR_TYPE__)dst,
                0, 0);
        }
        __builtin_amdgcn_s_wait_asynccnt(0);
#else
        u32x4* dst = (u32x4*)tile;
        for (int t = tid; t < TILE_COLS * QPC; t += 256) {
            int col = t / QPC, q = t % QPC;
            dst[t] = ((const u32x4*)(XB + (size_t)(n0 + col) * CI_P))[q];
        }
#endif
    }
    __syncthreads();

    const v8f zero8 = {0.f, 0.f, 0.f, 0.f, 0.f, 0.f, 0.f, 0.f};
    v8f accT[4] = {zero8, zero8, zero8, zero8};
    v8f accG[4] = {zero8, zero8, zero8, zero8};

    const int pad = (Kt - 1) >> 1;
    for (int k = 0; k < Kt; ++k) {
        const int d = k - pad;                          // tap offset
        const unsigned short* aT0 = WT + ((size_t)(m0 + l15) * Kt + k) * CI_P + lh * 8;
        const unsigned short* aG0 = WG + ((size_t)(m0 + l15) * Kt + k) * CI_P + lh * 8;
        for (int c0 = 0; c0 < CI_P; c0 += 32) {
            v16bf aT = load_frag_a(aT0 + c0);
            v16bf aG = load_frag_a(aG0 + c0);
#pragma unroll
            for (int ns = 0; ns < 4; ++ns) {
                const unsigned short* pb =
                    tile + (size_t)(4 + d + ns * 16 + l15) * CI_P + c0 + lh * 8;
                v16bf bf = load_frag_b(pb);
                accT[ns] = wmma_bf16(aT, bf, accT[ns]);
                accG[ns] = wmma_bf16(aG, bf, accG[ns]);
            }
        }
    }

    // epilogue: C/D layout -> element r is row m0 + 8*lh + r, col n0+ns*16+l15
    float btv[8], bgv[8];
#pragma unroll
    for (int r = 0; r < 8; ++r) {
        int co = m0 + 8 * lh + r;
        btv[r] = bt[co];
        bgv[r] = bg[co] + Evec[co];
    }
#pragma unroll
    for (int ns = 0; ns < 4; ++ns) {
        int l = n0 + ns * 16 + l15;
        unsigned short h[8];
#pragma unroll
        for (int r = 0; r < 8; ++r) {
            float tv = tanhf(accT[ns][r] + btv[r]);
            float gv = accG[ns][r] + bgv[r];
            gv = gv > 0.f ? gv : 0.f;
            h[r] = f2bf(tv * gv);
        }
        u32x4 o = { (unsigned)h[0] | ((unsigned)h[1] << 16),
                    (unsigned)h[2] | ((unsigned)h[3] << 16),
                    (unsigned)h[4] | ((unsigned)h[5] << 16),
                    (unsigned)h[6] | ((unsigned)h[7] << 16) };
        *(u32x4*)(Cout + (size_t)l * CONV + m0 + 8 * lh) = o;
    }
}

// ---------------------------------------------------------------------------
// Attention scores: scores[s][l] = relation . tanh(attW @ c_l + att_b)
// Block: 128 thr (4 waves), 16 cols per wave; WMMA over K=512, M chunks of 16.
// ---------------------------------------------------------------------------
__global__ __launch_bounds__(128)
void kAttScore(const unsigned short* __restrict__ AWB,   // [304][512] bf16
               const unsigned short* __restrict__ Cbuf,  // [6][4096][512] bf16
               const float* __restrict__ rel,
               const float* __restrict__ attb,
               float* __restrict__ scores) {
    __shared__ float sSc[64];
    const int s    = blockIdx.z;
    const int cB   = blockIdx.x * 64;
    const int tid  = threadIdx.x;
    const int wave = tid >> 5;
    const int lane = tid & 31;
    const int l15  = lane & 15;
    const int lh   = lane >> 4;
    const int colBase = cB + wave * 16;

    const unsigned short* cb = Cbuf + (size_t)s * SEN * CONV;
    const v8f zero8 = {0.f, 0.f, 0.f, 0.f, 0.f, 0.f, 0.f, 0.f};

    float colAcc = 0.f;
    for (int mc = 0; mc < 19; ++mc) {                   // 19*16 = 304 rows
        v8f acc = zero8;
        const unsigned short* aRow = AWB + (size_t)(mc * 16 + l15) * CONV + lh * 8;
        const unsigned short* bCol = cb + (size_t)(colBase + l15) * CONV + lh * 8;
        for (int kk = 0; kk < CONV; kk += 32) {
            v16bf a = load_frag_a(aRow + kk);
            v16bf b = load_frag_b(bCol + kk);
            acc = wmma_bf16(a, b, acc);
        }
#pragma unroll
        for (int r = 0; r < 8; ++r) {
            int m = mc * 16 + 8 * lh + r;
            if (m < VEC) colAcc += rel[m] * tanhf(acc[r] + attb[m]);
        }
    }
    // fold the two lane halves (same column) deterministically
    colAcc += __shfl_xor(colAcc, 16, 32);
    if (lane < 16) sSc[wave * 16 + l15] = colAcc;
    __syncthreads();
    if (tid < 64) scores[(size_t)s * SEN + cB + tid] = sSc[tid];
}

// ---------------------------------------------------------------------------
__global__ __launch_bounds__(256)
void kSoftmax(const float* __restrict__ scores, float* __restrict__ att) {
    __shared__ float red[256];
    const int s = blockIdx.x, tid = threadIdx.x;
    const float* sc = scores + (size_t)s * SEN;
    float* at = att + (size_t)s * SEN;
    float mx = -3.4e38f;
    for (int l = tid; l < SEN; l += 256) mx = fmaxf(mx, sc[l]);
    red[tid] = mx; __syncthreads();
    for (int o = 128; o > 0; o >>= 1) { if (tid < o) red[tid] = fmaxf(red[tid], red[tid + o]); __syncthreads(); }
    mx = red[0]; __syncthreads();
    float sum = 0.f;
    for (int l = tid; l < SEN; l += 256) { float e = __expf(sc[l] - mx); at[l] = e; sum += e; }
    red[tid] = sum; __syncthreads();
    for (int o = 128; o > 0; o >>= 1) { if (tid < o) red[tid] += red[tid + o]; __syncthreads(); }
    float inv = 1.f / red[0];
    __syncthreads();
    for (int l = tid; l < SEN; l += 256) at[l] *= inv;
}

// pooled[s][co] = sum_l c[s][l][co] * att[s][l]
__global__ __launch_bounds__(256)
void kPool(const unsigned short* __restrict__ Cbuf, const float* __restrict__ att,
           float* __restrict__ feat) {
    const int s  = blockIdx.z;
    const int co = blockIdx.x * 256 + threadIdx.x;
    const unsigned short* cb = Cbuf + (size_t)s * SEN * CONV + co;
    const float* at = att + (size_t)s * SEN;
    float acc = 0.f;
    for (int l = 0; l < SEN; ++l) acc += bf2f(cb[(size_t)l * CONV]) * at[l];
    feat[s * CONV + co] = acc;
}

__global__ __launch_bounds__(256)
void kFinal(const float* __restrict__ feat,
            const float* __restrict__ llW, const float* __restrict__ llb,
            const float* __restrict__ smW, const float* __restrict__ smb,
            float* __restrict__ out) {
    __shared__ float sF[3072];
    __shared__ float sH[CONV];
    __shared__ float red[256];
    const int tid = threadIdx.x;
    for (int j = tid; j < 3072; j += 256) sF[j] = feat[j];
    __syncthreads();
    for (int co = tid; co < CONV; co += 256) {
        float acc = llb[co];
        const float* row = llW + (size_t)co * 3072;
        for (int j = 0; j < 3072; ++j) acc += row[j] * sF[j];
        sH[co] = acc > 0.f ? acc : 0.f;
    }
    __syncthreads();
    for (int c = 0; c < 2; ++c) {
        float p = 0.f;
        for (int k = tid; k < CONV; k += 256) p += smW[c * CONV + k] * sH[k];
        red[tid] = p; __syncthreads();
        for (int o = 128; o > 0; o >>= 1) { if (tid < o) red[tid] += red[tid + o]; __syncthreads(); }
        if (tid == 0) out[c] = red[0] + smb[c];
        __syncthreads();
    }
}

// ---------------------------------------------------------------------------
// Workspace layout (bytes); total ~44.1 MB
// ---------------------------------------------------------------------------
static constexpr size_t OXB = 0;
static constexpr size_t SZXB = (size_t)XCOLS * CI_P * 2;
static constexpr size_t OWB = OXB + SZXB;
static constexpr size_t SZWB = (size_t)4 * CONV * 12 * CI_P * 2;
static constexpr size_t OAW = OWB + SZWB;
static constexpr size_t SZAW = (size_t)304 * CONV * 2;
static constexpr size_t OE  = OAW + SZAW;
static constexpr size_t SZE = 2 * CONV * 4;
static constexpr size_t OC  = OE + SZE;
static constexpr size_t SZC = (size_t)6 * SEN * CONV * 2;
static constexpr size_t OSC = OC + SZC;
static constexpr size_t SZSC = (size_t)6 * SEN * 4;
static constexpr size_t OAT = OSC + SZSC;
static constexpr size_t SZAT = (size_t)6 * SEN * 4;
static constexpr size_t OFT = OAT + SZAT;

static inline size_t convOffElems(int conv) {
    int g = conv / 3, b = conv % 3;
    int sk = (b == 0) ? 0 : ((b == 1) ? 3 : 7);
    return ((size_t)g * 12 + sk) * CONV * CI_P;
}

extern "C" void kernel_launch(void* const* d_in, const int* in_sizes, int n_in,
                              void* d_out, int out_size, void* d_ws, size_t ws_size,
                              hipStream_t stream) {
    const float* x    = (const float*)d_in[0];
    const float* e1   = (const float*)d_in[1];
    const float* e2   = (const float*)d_in[2];
    const float* rel  = (const float*)d_in[3];
    // convs: tc 4..9, c 10..15, td 16..21, d 22..27 (W,b interleaved)
    const float* chemW = (const float*)d_in[28];
    const float* chemB = (const float*)d_in[29];
    const float* disW  = (const float*)d_in[30];
    const float* disB  = (const float*)d_in[31];
    const float* attW  = (const float*)d_in[32];
    const float* attB  = (const float*)d_in[33];
    const float* llW   = (const float*)d_in[34];
    const float* llB   = (const float*)d_in[35];
    const float* smW   = (const float*)d_in[36];
    const float* smB   = (const float*)d_in[37];

    char* ws = (char*)d_ws;
    unsigned short* XB   = (unsigned short*)(ws + OXB);
    unsigned short* WB   = (unsigned short*)(ws + OWB);
    unsigned short* AWB  = (unsigned short*)(ws + OAW);
    float*          Ebuf = (float*)(ws + OE);
    unsigned short* Cbuf = (unsigned short*)(ws + OC);
    float*          scr  = (float*)(ws + OSC);
    float*          att  = (float*)(ws + OAT);
    float*          feat = (float*)(ws + OFT);

    // ---- prep ----
    kConvertX<<<(XCOLS * CI_P + 255) / 256, 256, 0, stream>>>(x, XB);

    WPtrs wp;
    for (int g = 0; g < 4; ++g)
        for (int b = 0; b < 3; ++b)
            wp.p[g * 3 + b] = (const float*)d_in[4 + g * 6 + b * 2];
    kConvertW<<<dim3((CONV * 5 * CI_P + 255) / 256, 1, 12), 256, 0, stream>>>(wp, WB);

    kConvertAttW<<<(304 * CONV + 255) / 256, 256, 0, stream>>>(attW, AWB);
    kEntGemv<<<4, 256, 0, stream>>>(chemW, chemB, disW, disB, e1, e2, Ebuf);

    // ---- fused conv+gate WMMA GEMMs, one launch per stream ----
    for (int s = 0; s < 6; ++s) {
        int b = s % 3, side = s / 3, Kt = 3 + b;
        int tConv = (side * 2) * 3 + b;
        int gConv = (side * 2 + 1) * 3 + b;
        const unsigned short* WT = WB + convOffElems(tConv);
        const unsigned short* WG = WB + convOffElems(gConv);
        const float* bt = (const float*)d_in[4 + (side * 2) * 6 + b * 2 + 1];
        const float* bg = (const float*)d_in[4 + (side * 2 + 1) * 6 + b * 2 + 1];
        kConvGemm<<<dim3(SEN / 64, CONV / 128), 256, 0, stream>>>(
            XB, WT, WG, bt, bg, Ebuf + side * CONV,
            Cbuf + (size_t)s * SEN * CONV, Kt);
    }

    // ---- attention: WMMA scores, softmax, pooling ----
    kAttScore<<<dim3(SEN / 64, 1, 6), 128, 0, stream>>>(AWB, Cbuf, rel, attB, scr);
    kSoftmax<<<6, 256, 0, stream>>>(scr, att);
    kPool<<<dim3(CONV / 256, 1, 6), 256, 0, stream>>>(Cbuf, att, feat);

    // ---- final MLP ----
    kFinal<<<1, 256, 0, stream>>>(feat, llW, llB, smW, smB, (float*)d_out);
}
